// CommonSpace2Edge_56324201119777
// MI455X (gfx1250) — matrix operand order repaired and checked by reading, not compiled, not run on hardware
//
#include <hip/hip_runtime.h>

// ---------------- problem constants ----------------
#define NE   131072      // edges
#define CD   256         // common_space (K)
#define DD   256         // edge_dim (N of gemm)
#define NTY  8           // edge types
#define RPT  128         // rows (edges) per gemm workgroup tile
#define MAXT (NE/RPT + NTY)   // upper bound on single-type tiles = 1032
#define LDSW 264         // 256 halves + 8 pad -> conflict-free ds_load_b128

typedef _Float16 v16h __attribute__((ext_vector_type(16)));
typedef _Float16 v8h  __attribute__((ext_vector_type(8)));
typedef _Float16 v4h  __attribute__((ext_vector_type(4)));
typedef float    v8f  __attribute__((ext_vector_type(8)));

__device__ __forceinline__ int imin(int a, int b){ return a < b ? a : b; }

// ---------------- workspace layout (bytes) ----------------
#define WS_WH   0                           // fp16 weights (T,D,C): 1 MiB
#define WS_PERM (WS_WH   + NTY*DD*CD*2)     // perm: N ints (512 KiB)
#define WS_HIST (WS_PERM + NE*4)
#define WS_CUR  (WS_HIST + 32)
#define WS_NTL  (WS_CUR  + 32)
#define WS_TILE (WS_NTL  + 16)              // MAXT * int4
// total ~1.59 MiB

// ---------------- tiny pipeline kernels ----------------
__global__ void k_init(int* __restrict__ hist){
  if (threadIdx.x < NTY) hist[threadIdx.x] = 0;
}

__global__ void k_wconv(const float* __restrict__ w, _Float16* __restrict__ wh){
  int i = blockIdx.x*256 + threadIdx.x;        // float4 index; grid covers T*D*C/4
  float4 v = ((const float4*)w)[i];
  v4h h; h[0]=(_Float16)v.x; h[1]=(_Float16)v.y; h[2]=(_Float16)v.z; h[3]=(_Float16)v.w;
  ((v4h*)wh)[i] = h;
}

__global__ void k_hist(const int* __restrict__ ty, int* __restrict__ hist){
  __shared__ int lh[NTY];
  int tid = threadIdx.x;
  if (tid < NTY) lh[tid] = 0;
  __syncthreads();
  atomicAdd(&lh[ty[blockIdx.x*256 + tid]], 1);
  __syncthreads();
  if (tid < NTY && lh[tid]) atomicAdd(&hist[tid], lh[tid]);
}

__global__ void k_scan(const int* __restrict__ hist, int* __restrict__ cur,
                       int4* __restrict__ tiles, int* __restrict__ ntiles){
  if (threadIdx.x != 0) return;
  int off[NTY]; int run = 0;
  for (int t = 0; t < NTY; ++t){ off[t] = run; run += hist[t]; }
  for (int t = 0; t < NTY; ++t) cur[t] = off[t];
  int n = 0;
  for (int t = 0; t < NTY; ++t){
    int c = hist[t];
    for (int s = 0; s < c; s += RPT){
      tiles[n] = make_int4(t, off[t] + s, imin(RPT, c - s), 0);
      ++n;
    }
  }
  *ntiles = n;
}

__global__ void k_scatter(const int* __restrict__ ty, int* __restrict__ cur,
                          int* __restrict__ perm){
  __shared__ int lh[NTY];
  __shared__ int lbase[NTY];
  int tid = threadIdx.x;
  if (tid < NTY) lh[tid] = 0;
  __syncthreads();
  int i = blockIdx.x*256 + tid;
  int t = ty[i];
  atomicAdd(&lh[t], 1);
  __syncthreads();
  if (tid < NTY){ lbase[tid] = atomicAdd(&cur[tid], lh[tid]); lh[tid] = 0; }
  __syncthreads();
  int slot = lbase[t] + atomicAdd(&lh[t], 1);
  perm[slot] = i;
}

// ---------------- grouped GEMM: 128 same-type rows x 256 outputs per WG ----------------
__global__ __launch_bounds__(256) void k_gemm(
    const float* __restrict__ x, const _Float16* __restrict__ wh,
    const float* __restrict__ bias, const int* __restrict__ perm,
    const int4* __restrict__ tiles, const int* __restrict__ ntiles,
    float* __restrict__ out){
  __shared__ _Float16 xs[RPT*LDSW];          // 66 KiB staged fp16 activations
  __shared__ __align__(16) int sperm[RPT];

  int b = blockIdx.x;
  if (b >= *ntiles) return;
  int4 tt = tiles[b];
  int t = tt.x, seg = tt.y, valid = tt.z;   // valid in [1,128]
  int tid = threadIdx.x;

  if (tid < RPT) sperm[tid] = perm[seg + imin(tid, valid - 1)];
  __syncthreads();

  // gather x rows (coalesced 1KB/row), convert fp32 -> fp16, land in LDS
  for (int i = tid; i < RPT*64; i += 256){
    int r = i >> 6, c4 = i & 63;
    float4 v = ((const float4*)x)[(unsigned)(sperm[r]*64 + c4)];
    v4h h; h[0]=(_Float16)v.x; h[1]=(_Float16)v.y; h[2]=(_Float16)v.z; h[3]=(_Float16)v.w;
    *(v4h*)&xs[r*LDSW + c4*4] = h;
  }
  __syncthreads();

  int lane = tid & 31, wave = tid >> 5;
  int l16 = lane & 15, lhi = lane >> 4;   // B/D column = l16; K-half select = lhi

  for (int cti = 0; cti < 2; ++cti){
    int n = (wave*2 + cti)*16 + l16;                 // this lane's output column
    const _Float16* wrow = wh + ((size_t)t*DD + n)*CD;
    // B panel for the whole K=256: 8 fragments in registers, reused by 8 row tiles.
    // B layout (32x16 f16): lane l16 = column, 16 contiguous K at kc*32 + 16*lhi.
    v16h B[8];
    #pragma unroll
    for (int kc = 0; kc < 8; ++kc)
      B[kc] = *(const v16h*)(wrow + kc*32 + lhi*16);
    float bv = bias[t*DD + n];

    for (int rt = 0; rt < 8; ++rt){
      v8f acc = {0.f,0.f,0.f,0.f,0.f,0.f,0.f,0.f};
      const _Float16* arow = &xs[(rt*16 + l16)*LDSW + lhi*16];
      #pragma unroll
      for (int kc = 0; kc < 8; ++kc){
        // Scheduling fence: keep each A-fragment LDS load glued to its WMMA so
        // the scheduler cannot hoist all 16 ds_loads and spill to scratch.
        asm volatile("" ::: "memory");
        // A layout (16x32 f16): lane row = l16; halves 0-7 at K = kc*32 + 16*lhi,
        // halves 8-15 at +8.  Two 16B LDS reads, bank-conflict-free via row pad.
        v8h alo = *(const v8h*)(arow + kc*32);
        v8h ahi = *(const v8h*)(arow + kc*32 + 8);
        v16h a = __builtin_shufflevector(alo, ahi,
                   0,1,2,3,4,5,6,7,8,9,10,11,12,13,14,15);
        acc = __builtin_amdgcn_wmma_f32_16x16x32_f16(false, a, false, B[kc],
                                                     (short)0, acc, false, false);
      }
      // D layout: VGPR r -> M = rt*16 + r + 8*lhi, N = l16.
      // Row indices for this lane-half are 8 contiguous ints: fetch as 2x b128.
      int rbase = rt*16 + lhi*8;
      int4 ra = *(const int4*)&sperm[rbase];
      int4 rb = *(const int4*)&sperm[rbase + 4];
      int rows[8] = {ra.x, ra.y, ra.z, ra.w, rb.x, rb.y, rb.z, rb.w};
      if (valid == RPT){
        // fast path: full tile, branch-free stores; 32-bit element offsets so the
        // compiler can use saddr + scaled-voffset stores (no 64-bit VGPR adds).
        #pragma unroll
        for (int r = 0; r < 8; ++r)
          out[(unsigned)(rows[r]*DD + n)] = acc[r] + bv;
      } else {
        #pragma unroll
        for (int r = 0; r < 8; ++r){
          if (rbase + r < valid)
            out[(unsigned)(rows[r]*DD + n)] = acc[r] + bv;
        }
      }
    }
  }
}

// ---------------- launch ----------------
extern "C" void kernel_launch(void* const* d_in, const int* in_sizes, int n_in,
                              void* d_out, int out_size, void* d_ws, size_t ws_size,
                              hipStream_t stream){
  (void)in_sizes; (void)n_in; (void)out_size; (void)ws_size;
  const float* x  = (const float*)d_in[0];   // [N, C]
  const float* w  = (const float*)d_in[1];   // [T, D, C]
  const float* bs = (const float*)d_in[2];   // [T, D]
  const int*   ty = (const int*)d_in[3];     // [N]
  float* out = (float*)d_out;                // [N, D]
  char* ws = (char*)d_ws;

  _Float16* wh  = (_Float16*)(ws + WS_WH);
  int*  perm    = (int*) (ws + WS_PERM);
  int*  hist    = (int*) (ws + WS_HIST);
  int*  cur     = (int*) (ws + WS_CUR);
  int*  ntl     = (int*) (ws + WS_NTL);
  int4* tiles   = (int4*)(ws + WS_TILE);

  k_init   <<<1, 32, 0, stream>>>(hist);
  k_wconv  <<<(NTY*DD*CD/4)/256, 256, 0, stream>>>(w, wh);
  k_hist   <<<NE/256, 256, 0, stream>>>(ty, hist);
  k_scan   <<<1, 1, 0, stream>>>(hist, cur, tiles, ntl);
  k_scatter<<<NE/256, 256, 0, stream>>>(ty, cur, perm);
  k_gemm   <<<MAXT, 256, 0, stream>>>(x, wh, bs, perm, tiles, ntl, out);
}
